// Model_58145267253421
// MI455X (gfx1250) — matrix operand order repaired and checked by reading, not compiled, not run on hardware
//
#include <hip/hip_runtime.h>
#include <math.h>

// ---------------- types for WMMA / TDM ----------------
typedef __attribute__((ext_vector_type(16))) __bf16 v16bf;
typedef __attribute__((ext_vector_type(8)))  float  v8f;
typedef __attribute__((ext_vector_type(8)))  unsigned int v8u;
typedef __attribute__((ext_vector_type(4)))  unsigned int u32x4;
typedef __attribute__((ext_vector_type(8)))  int i32x8;
typedef __attribute__((ext_vector_type(4)))  int i32x4;

#if __has_builtin(__builtin_amdgcn_tensor_load_to_lds)
#define USE_TDM 1
#else
#define USE_TDM 0
#endif

#define LDS_STRIDE 516   // 512 data dwords + 4 pad dwords (TDM pad_interval=7, pad_amount=3)

// ---------------- bf16 helpers ----------------
__device__ __forceinline__ float bf2f(unsigned int h) {
    unsigned int u = h << 16;
    float f;
    __builtin_memcpy(&f, &u, 4);
    return f;
}
__device__ __forceinline__ unsigned short f2bf(float f) {
    unsigned int u;
    __builtin_memcpy(&u, &f, 4);
    u += 0x7fffu + ((u >> 16) & 1u);   // round-to-nearest-even
    return (unsigned short)(u >> 16);
}
__device__ __forceinline__ unsigned int packbf(float a, float b) {
    return (unsigned int)f2bf(a) | ((unsigned int)f2bf(b) << 16);
}

#if USE_TDM
// 2-D TDM: load [rows] x [512 dwords] tile (row stride M dwords) into LDS with
// 4-dword padding after every 512 dwords (LDS row stride = 516 dwords).
__device__ __forceinline__ void tdm_load_tile(unsigned ldsAddr, const unsigned int* gptr,
                                              int rows, int M) {
    unsigned long long ga = (unsigned long long)gptr;
    u32x4 g0;
    g0.x = 1u;                                               // count=1
    g0.y = ldsAddr;                                          // lds_addr (bytes)
    g0.z = (unsigned)ga;                                     // global_addr[31:0]
    g0.w = (unsigned)((ga >> 32) & 0x1ffffffull) | (2u << 30); // global_addr[56:32], type=2
    i32x8 g1;
    g1[0] = (int)((2u << 16) | (1u << 20) | (7u << 22) | (3u << 25)); // 4B elems, pad en, 512dw/4dw
    g1[1] = (int)(((unsigned)M & 0xffffu) << 16);                     // tensor_dim0 lo
    g1[2] = (int)((((unsigned)M >> 16) & 0xffffu) |
                  (((unsigned)rows & 0xffffu) << 16));                // tensor_dim0 hi | tensor_dim1 lo
    g1[3] = (int)(512u << 16);                                        // tensor_dim1 hi | tile_dim0=512
    g1[4] = (int)((unsigned)rows & 0xffffu);                          // tile_dim1, tile_dim2=0
    g1[5] = (int)(unsigned)M;                                         // tensor_dim0_stride lo
    g1[6] = 0;
    g1[7] = 0;
    i32x4 z4 = {0, 0, 0, 0};
#if defined(__clang_major__) && (__clang_major__ >= 23)
    i32x8 z8 = {0, 0, 0, 0, 0, 0, 0, 0};
    __builtin_amdgcn_tensor_load_to_lds(g0, g1, z4, z4, z8, 0);
#else
    __builtin_amdgcn_tensor_load_to_lds(g0, g1, z4, z4, 0);
#endif
}
#endif

// ---------------- utility ----------------
__global__ void zero_f32(float* p, int n) {
    int i = blockIdx.x * 256 + threadIdx.x;
    if (i < n) p[i] = 0.f;
}

// Pack f32 weights (Cout,Cin) into WMMA A-fragment layout (bf16 pairs), zero-padded.
// Layout: Wp[((mt*Kt + kt)*8 + v)*32 + lane]
__global__ void pack_w(const float* __restrict__ w, unsigned int* __restrict__ Wp,
                       int Cout, int Cin, int Kt) {
    int mt = blockIdx.x / Kt, kt = blockIdx.x % Kt;
    int t = threadIdx.x;
    int lane = t & 31, v = t >> 5;
    int m = mt * 16 + (lane & 15);
    int hi = (lane >= 16) ? 16 : 0;
    int k0 = kt * 32 + hi + 2 * v;
    float a = (m < Cout && k0 < Cin) ? w[(size_t)m * Cin + k0] : 0.f;
    float b = (m < Cout && (k0 + 1) < Cin) ? w[(size_t)m * Cin + k0 + 1] : 0.f;
    Wp[((size_t)blockIdx.x * 8 + v) * 32 + lane] = packbf(a, b);
}

// ---------------- WMMA 1x1-conv GEMM ----------------
// Block owns 512 positions; X tile staged ONCE in LDS (TDM DMA); loops all M-tiles.
// Y raw bf16 out + fused BN sum/sumsq stats.
__global__ void __launch_bounds__(256)
wmma_conv(const unsigned int* __restrict__ Wp, const unsigned int* __restrict__ Xp,
          unsigned int* __restrict__ Yp, float* __restrict__ stats,
          int M, int Kt, int Cout) {
    extern __shared__ unsigned int ldsX[];
    const int lane = threadIdx.x & 31;
    const int wave = threadIdx.x >> 5;
    const int tile0 = blockIdx.x * 512;
    const int rows = Kt * 16;   // bf16-pair rows

#if USE_TDM
    if (threadIdx.x == 0) {
        tdm_load_tile((unsigned)(unsigned long long)(void*)&ldsX[0], Xp + tile0, rows, M);
    }
    __builtin_amdgcn_s_wait_tensorcnt(0);
#else
    for (int idx = threadIdx.x; idx < rows * 512; idx += 256) {
        int r = idx >> 9, c = idx & 511;
        int col = tile0 + c;
        ldsX[r * LDS_STRIDE + c] = (col < M) ? Xp[(size_t)r * M + col] : 0u;
    }
#endif
    __syncthreads();

    const int n0 = wave * 64;            // local position base of this wave
    if (tile0 + n0 >= M) return;         // wave-uniform; EXEC stays all-ones below
    const int half = (lane >> 4) & 1;
    const int ln = lane & 15;
    const int Mt = Cout >> 4;

    for (int mt = 0; mt < Mt; ++mt) {
        v8f acc[4];
#pragma unroll
        for (int s = 0; s < 4; ++s) acc[s] = v8f{};

        const unsigned int* wptr = Wp + (size_t)(mt * Kt) * 256 + lane;
        for (int kt = 0; kt < Kt; ++kt) {
            v8u au;
#pragma unroll
            for (int v = 0; v < 8; ++v) au[v] = wptr[v * 32];
            wptr += 256;
            v16bf a = __builtin_bit_cast(v16bf, au);
            const unsigned int* xrow = ldsX + (kt * 16 + half * 8) * LDS_STRIDE + n0 + ln;
#pragma unroll
            for (int sub = 0; sub < 4; ++sub) {
                v8u bu;
#pragma unroll
                for (int j = 0; j < 8; ++j) bu[j] = xrow[j * LDS_STRIDE + sub * 16];
                v16bf b = __builtin_bit_cast(v16bf, bu);
                acc[sub] = __builtin_amdgcn_wmma_f32_16x16x32_bf16(
                    false, a, false, b, (short)0, acc[sub], false, false);
            }
        }

        float ssum[8], ssq[8];
#pragma unroll
        for (int e = 0; e < 8; ++e) { ssum[e] = 0.f; ssq[e] = 0.f; }
#pragma unroll
        for (int sub = 0; sub < 4; ++sub) {
            int n = tile0 + n0 + sub * 16 + ln;
#pragma unroll
            for (int r = 0; r < 4; ++r) {
                float a0 = acc[sub][2 * r], a1 = acc[sub][2 * r + 1];
                Yp[(size_t)(mt * 8 + half * 4 + r) * M + n] = packbf(a0, a1);
            }
#pragma unroll
            for (int e = 0; e < 8; ++e) { float v = acc[sub][e]; ssum[e] += v; ssq[e] += v * v; }
        }
#pragma unroll
        for (int e = 0; e < 8; ++e) {
            float s = ssum[e], q = ssq[e];
#pragma unroll
            for (int off = 1; off < 16; off <<= 1) {
                s += __shfl_xor(s, off, 32);
                q += __shfl_xor(q, off, 32);
            }
            if (ln == 0) {
                int c = mt * 16 + half * 8 + e;
                atomicAdd(&stats[c], s);
                atomicAdd(&stats[Cout + c], q);
            }
        }
    }
}

// ---------------- BN ----------------
__global__ void bn_fin(const float* __restrict__ stats, const float* __restrict__ g,
                       const float* __restrict__ be, float* __restrict__ sc,
                       float* __restrict__ sh, int C, float Mf) {
    int c = threadIdx.x;
    if (c >= C) return;
    float mean = stats[c] / Mf;
    float var = stats[C + c] / Mf - mean * mean;
    float s = g[c] * rsqrtf(var + 1e-5f);
    sc[c] = s;
    sh[c] = be[c] - mean * s;
}

__global__ void bn_apply(const unsigned int* __restrict__ src, const unsigned int* __restrict__ res,
                         unsigned int* __restrict__ dst, const float* __restrict__ sc,
                         const float* __restrict__ sh, int pairs, int M, int relu) {
    size_t idx = (size_t)blockIdx.x * 256 + threadIdx.x;
    size_t total = (size_t)pairs * M;
    if (idx >= total) return;
    int p = (int)(idx / M);
    int c0 = 2 * p;
    unsigned int d = src[idx];
    float v0 = bf2f(d & 0xffffu) * sc[c0] + sh[c0];
    float v1 = bf2f(d >> 16) * sc[c0 + 1] + sh[c0 + 1];
    if (res) {
        unsigned int r = res[idx];
        v0 += bf2f(r & 0xffffu);
        v1 += bf2f(r >> 16);
    }
    if (relu) { v0 = fmaxf(v0, 0.f); v1 = fmaxf(v1, 0.f); }
    dst[idx] = packbf(v0, v1);
}

// ---------------- embed (Cin=2 -> 64), bias dropped (BN-invariant) ----------------
__global__ void __launch_bounds__(256)
embed_kernel(const float* __restrict__ x, const float* __restrict__ w,
             unsigned int* __restrict__ Yp, float* __restrict__ stats) {
    int m = blockIdx.x * 256 + threadIdx.x;   // < 32000
    int lane = threadIdx.x & 31;
    float x0 = x[2 * m], x1 = x[2 * m + 1];
#pragma unroll 4
    for (int p = 0; p < 32; ++p) {
        float y0 = w[4 * p] * x0 + w[4 * p + 1] * x1;
        float y1 = w[4 * p + 2] * x0 + w[4 * p + 3] * x1;
        Yp[p * 32000 + m] = packbf(y0, y1);
        float s0 = y0, q0 = y0 * y0, s1 = y1, q1 = y1 * y1;
        for (int off = 1; off < 32; off <<= 1) {
            s0 += __shfl_xor(s0, off, 32); q0 += __shfl_xor(q0, off, 32);
            s1 += __shfl_xor(s1, off, 32); q1 += __shfl_xor(q1, off, 32);
        }
        if (lane == 0) {
            atomicAdd(&stats[2 * p], s0);      atomicAdd(&stats[64 + 2 * p], q0);
            atomicAdd(&stats[2 * p + 1], s1);  atomicAdd(&stats[64 + 2 * p + 1], q1);
        }
    }
}

// ---------------- farthest point sampling (serial over S; one block per batch) ----------------
__global__ void __launch_bounds__(256)
fps_kernel(const float* __restrict__ xy, int* __restrict__ sidx, float* __restrict__ nxy,
           int Npts, int S) {
    __shared__ float dist[2000];
    __shared__ float rv[256];
    __shared__ int ri[256];
    __shared__ float cc[2];
    int b = blockIdx.x, t = threadIdx.x;
    const float* bxy = xy + (size_t)b * Npts * 2;
    for (int n = t; n < Npts; n += 256) dist[n] = 3.4e38f;
    int far = 0;
    __syncthreads();
    for (int s = 0; s < S; ++s) {
        if (t == 0) {
            sidx[b * S + s] = far;
            cc[0] = bxy[2 * far]; cc[1] = bxy[2 * far + 1];
            nxy[(size_t)(b * S + s) * 2] = cc[0];
            nxy[(size_t)(b * S + s) * 2 + 1] = cc[1];
        }
        __syncthreads();
        float cx = cc[0], cy = cc[1];
        float best = -1.f; int bi = 0;
        for (int n = t; n < Npts; n += 256) {
            float dx = bxy[2 * n] - cx, dy = bxy[2 * n + 1] - cy;
            float dn = fminf(dist[n], dx * dx + dy * dy);
            dist[n] = dn;
            if (dn > best) { best = dn; bi = n; }
        }
        rv[t] = best; ri[t] = bi;
        __syncthreads();
        for (int off = 128; off > 0; off >>= 1) {
            if (t < off) {
                if (rv[t + off] > rv[t] || (rv[t + off] == rv[t] && ri[t + off] < ri[t])) {
                    rv[t] = rv[t + off]; ri[t] = ri[t + off];
                }
            }
            __syncthreads();
        }
        far = ri[0];
        __syncthreads();
    }
}

// ---------------- KNN + gather + center + batch-std stats (one block per (b,s)) ----------------
__global__ void __launch_bounds__(256)
knn_group(const float* __restrict__ xy, const float* __restrict__ nxy,
          const unsigned int* __restrict__ featP, unsigned short* __restrict__ cent,
          float* __restrict__ bstats, int Npts, int S, int C, int featM, int Mg) {
    __shared__ float dist[2000];
    __shared__ int nbr[64];
    __shared__ float rv[256];
    __shared__ int ri[256];
    int bs = blockIdx.x, b = bs / S, t = threadIdx.x;
    const float* bxy = xy + (size_t)b * Npts * 2;
    float cx = nxy[(size_t)bs * 2], cy = nxy[(size_t)bs * 2 + 1];
    for (int n = t; n < Npts; n += 256) {
        float dx = bxy[2 * n] - cx, dy = bxy[2 * n + 1] - cy;
        dist[n] = dx * dx + dy * dy;
    }
    __syncthreads();
    for (int k = 0; k < 64; ++k) {
        float best = 3.4e38f; int bi = 0;
        for (int n = t; n < Npts; n += 256) {
            float d = dist[n];
            if (d < best || (d == best && n < bi)) { best = d; bi = n; }
        }
        rv[t] = best; ri[t] = bi;
        __syncthreads();
        for (int off = 128; off > 0; off >>= 1) {
            if (t < off) {
                if (rv[t + off] < rv[t] || (rv[t + off] == rv[t] && ri[t + off] < ri[t])) {
                    rv[t] = rv[t + off]; ri[t] = ri[t + off];
                }
            }
            __syncthreads();
        }
        if (t == 0) { nbr[k] = ri[0]; dist[ri[0]] = 3.4e38f; }
        __syncthreads();
    }
    int D = C + 2;
    float s1 = 0.f, s2 = 0.f;
    if (t < D) {
        int c = t;
        float mean = 0.f;
        for (int k = 0; k < 64; ++k) {
            int n = nbr[k];
            float v;
            if (c < C) {
                unsigned int dw = featP[(size_t)(c >> 1) * featM + b * Npts + n];
                v = bf2f((c & 1) ? (dw >> 16) : (dw & 0xffffu));
            } else v = bxy[2 * n + (c - C)];
            mean += v;
        }
        mean *= (1.0f / 64.0f);
        size_t base = (size_t)c * Mg + (size_t)bs * 64;
        for (int k = 0; k < 64; ++k) {
            int n = nbr[k];
            float v;
            if (c < C) {
                unsigned int dw = featP[(size_t)(c >> 1) * featM + b * Npts + n];
                v = bf2f((c & 1) ? (dw >> 16) : (dw & 0xffffu));
            } else v = bxy[2 * n + (c - C)];
            v -= mean;
            cent[base + k] = f2bf(v);
            s1 += v; s2 += v * v;
        }
    }
    rv[t] = s1; __syncthreads();
    for (int off = 128; off > 0; off >>= 1) { if (t < off) rv[t] += rv[t + off]; __syncthreads(); }
    float S1 = rv[0]; __syncthreads();
    rv[t] = s2; __syncthreads();
    for (int off = 128; off > 0; off >>= 1) { if (t < off) rv[t] += rv[t + off]; __syncthreads(); }
    if (t == 0) { atomicAdd(&bstats[2 * b], S1); atomicAdd(&bstats[2 * b + 1], rv[0]); }
}

__global__ void bstd_fin(const float* __restrict__ bstats, float* __restrict__ rstd, float n) {
    int b = threadIdx.x;
    if (b >= 16) return;
    float s = bstats[2 * b], q = bstats[2 * b + 1];
    float fm = s / n;
    float var = (q - n * fm * fm) / (n - 1.0f);
    rstd[b] = 1.0f / (sqrtf(var) + 1e-5f);
}

// Build padded transfer-GEMM input: [gnorm(C+2) | rep(C) | zero-pad] as bf16 pairs.
__global__ void group_scale(const unsigned short* __restrict__ cent, const float* __restrict__ rstd,
                            const float* __restrict__ alpha, const float* __restrict__ beta,
                            const unsigned int* __restrict__ featP, const int* __restrict__ sidx,
                            unsigned int* __restrict__ Xg, int C, int CpadPairs, int Mg,
                            int S, int featM, int Npts) {
    size_t idx = (size_t)blockIdx.x * 256 + threadIdx.x;
    size_t total = (size_t)CpadPairs * Mg;
    if (idx >= total) return;
    int p = (int)(idx / Mg);
    int m = (int)(idx % Mg);
    int c0 = 2 * p;
    int b = m / (S * 64);
    unsigned int out = 0u;
    if (c0 < C + 2) {
        float r = rstd[b];
        float v0 = bf2f(cent[(size_t)c0 * Mg + m]);
        float v1 = bf2f(cent[(size_t)(c0 + 1) * Mg + m]);
        v0 = alpha[c0] * (v0 * r) + beta[c0];
        v1 = alpha[c0 + 1] * (v1 * r) + beta[c0 + 1];
        out = packbf(v0, v1);
    } else if (c0 < 2 * C + 2) {
        int q = (c0 - (C + 2)) >> 1;
        int s = (m >> 6) % S;
        int sv = sidx[b * S + s];
        out = featP[(size_t)q * featM + b * Npts + sv];
    }
    Xg[idx] = out;
}

// ---------------- max-pool over K=64 ----------------
__global__ void maxpool_kernel(const unsigned int* __restrict__ in, unsigned int* __restrict__ out,
                               int pairs, int Mout) {
    size_t idx = (size_t)blockIdx.x * 256 + threadIdx.x;
    if (idx >= (size_t)pairs * Mout) return;
    int p = (int)(idx / Mout), m = (int)(idx % Mout);
    const unsigned int* src = in + (size_t)p * Mout * 64 + (size_t)m * 64;
    float m0 = -3.4e38f, m1 = -3.4e38f;
    for (int k = 0; k < 64; ++k) {
        unsigned int d = src[k];
        m0 = fmaxf(m0, bf2f(d & 0xffffu));
        m1 = fmaxf(m1, bf2f(d >> 16));
    }
    out[(size_t)p * Mout + m] = packbf(m0, m1);
}

// ---------------- final unpack to f32 (B,256,500) ----------------
__global__ void unpack_out(const unsigned int* __restrict__ p2, float* __restrict__ out) {
    int idx = blockIdx.x * 256 + threadIdx.x;
    if (idx >= 16 * 256 * 500) return;
    int s = idx % 500;
    int c = (idx / 500) % 256;
    int b = idx / (500 * 256);
    unsigned int d = p2[(size_t)(c >> 1) * 8000 + b * 500 + s];
    out[idx] = bf2f((c & 1) ? (d >> 16) : (d & 0xffffu));
}

// =========================================================================
extern "C" void kernel_launch(void* const* d_in, const int* in_sizes, int n_in,
                              void* d_out, int out_size, void* d_ws, size_t ws_size,
                              hipStream_t stream) {
    (void)in_sizes; (void)n_in; (void)out_size; (void)ws_size;
    const float* x = (const float*)d_in[0];

    const int Mg1 = 16 * 1000 * 64, Mg2 = 16 * 500 * 64;
    const int M0 = 32000, M1 = 16000, M2 = 8000;

    size_t cur = 0;
    auto alloc = [&](size_t bytes) -> void* {
        void* p = (void*)((char*)d_ws + cur);
        cur += (bytes + 255) & ~(size_t)255;
        return p;
    };
    unsigned int* raw0   = (unsigned int*)alloc((size_t)32 * M0 * 4);
    unsigned int* feat0  = (unsigned int*)alloc((size_t)32 * M0 * 4);
    float* statsBuf = (float*)alloc(512 * 4);
    float* scaleBuf = (float*)alloc(256 * 4);
    float* shiftBuf = (float*)alloc(256 * 4);
    int*   sidx1 = (int*)alloc(16000 * 4);
    float* nxy1  = (float*)alloc(32000 * 4);
    int*   sidx2 = (int*)alloc(8000 * 4);
    float* nxy2  = (float*)alloc(8000 * 2 * 4);
    float* bstats = (float*)alloc(32 * 4);
    float* rstdB  = (float*)alloc(16 * 4);
    unsigned short* cent = (unsigned short*)alloc((size_t)66 * Mg1 * 2);   // covers 130*Mg2 too
    unsigned int* Xg     = (unsigned int*)alloc((size_t)80 * Mg1 * 4);     // covers 144*Mg2
    unsigned int* bufRaw = (unsigned int*)alloc((size_t)64 * Mg1 * 4);     // == 128*Mg2
    unsigned int* bufX   = (unsigned int*)alloc((size_t)64 * Mg1 * 4);
    unsigned int* bufH   = (unsigned int*)alloc((size_t)64 * Mg1 * 4);
    unsigned int* p1     = (unsigned int*)alloc((size_t)64 * M1 * 4);
    unsigned int* p2     = (unsigned int*)alloc((size_t)128 * M2 * 4);
    unsigned int* posRaw = (unsigned int*)alloc((size_t)128 * M2 * 4);
    unsigned int* posH   = (unsigned int*)alloc((size_t)128 * M2 * 4);

    auto allocW = [&](int Mt, int Kt) { return (unsigned int*)alloc((size_t)Mt * Kt * 256 * 4); };
    unsigned int* wpT1 = allocW(8, 5);
    unsigned int* wpT2 = allocW(16, 9);
    unsigned int* wpS0[8];
    unsigned int* wpS1[8];
    for (int i = 0; i < 8; ++i) wpS0[i] = allocW(8, 4);
    for (int i = 0; i < 8; ++i) wpS1[i] = allocW(16, 8);

    auto zeroF = [&](float* p, int n) {
        zero_f32<<<dim3((n + 255) / 256), dim3(256), 0, stream>>>(p, n);
    };
    auto pack = [&](int wIdx, int Cout, int Cin, int Kt, unsigned int* Wp) {
        pack_w<<<dim3((Cout / 16) * Kt), dim3(256), 0, stream>>>(
            (const float*)d_in[wIdx], Wp, Cout, Cin, Kt);
    };
    // conv + BN(+residual)(+relu)
    auto conv = [&](const unsigned int* Wp, const unsigned int* X, unsigned int* Yraw,
                    int Cout, int Kt, int M, int gIdx, int beIdx,
                    const unsigned int* resid, unsigned int* dst, int relu) {
        zeroF(statsBuf, 2 * Cout);
        size_t ldsB = (size_t)Kt * 16 * LDS_STRIDE * 4;
        wmma_conv<<<dim3((unsigned)((M + 511) / 512)), dim3(256), ldsB, stream>>>(
            Wp, X, Yraw, statsBuf, M, Kt, Cout);
        bn_fin<<<dim3(1), dim3(256), 0, stream>>>(
            statsBuf, (const float*)d_in[gIdx], (const float*)d_in[beIdx],
            scaleBuf, shiftBuf, Cout, (float)M);
        size_t tot = (size_t)(Cout / 2) * M;
        bn_apply<<<dim3((unsigned)((tot + 255) / 256)), dim3(256), 0, stream>>>(
            Yraw, resid, dst, scaleBuf, shiftBuf, Cout / 2, M, relu);
    };
    auto resblock = [&](unsigned int* xbuf, unsigned int* rawb, unsigned int* hbuf,
                        int Cch, int Kt, int M, int n1base, int n2base,
                        const unsigned int* wp1, const unsigned int* wp2) {
        conv(wp1, xbuf, rawb, Cch, Kt, M, n1base + 2, n1base + 1, nullptr, hbuf, 1);
        conv(wp2, hbuf, rawb, Cch, Kt, M, n2base + 2, n2base + 1, xbuf, xbuf, 1);
    };

    // ---- pack all conv weights (d_in flattening: sorted keys, depth-first) ----
    // stage0: alpha=5 beta=6; pos0n1@7 pos0n2@11 pos1n1@15 pos1n2@19; pre0n1@23 pre0n2@27 pre1n1@31 pre1n2@35; transfer@39
    // stage1: alpha=43 beta=44; pos0n1@45 pos0n2@49 pos1n1@53 pos1n2@57; pre0n1@61 pre0n2@65 pre1n1@69 pre1n2@73; transfer@77
    const int w0[8] = {10, 14, 18, 22, 26, 30, 34, 38};
    const int w1[8] = {48, 52, 56, 60, 64, 68, 72, 76};
    pack(42, 128, 130, 5, wpT1);
    pack(80, 256, 258, 9, wpT2);
    for (int i = 0; i < 8; ++i) pack(w0[i], 128, 128, 4, wpS0[i]);
    for (int i = 0; i < 8; ++i) pack(w1[i], 256, 256, 8, wpS1[i]);

    // ---- embed: 2 -> 64 + BN + ReLU ----
    zeroF(statsBuf, 128);
    embed_kernel<<<dim3(125), dim3(256), 0, stream>>>(x, (const float*)d_in[4], raw0, statsBuf);
    bn_fin<<<dim3(1), dim3(256), 0, stream>>>(statsBuf, (const float*)d_in[3],
                                              (const float*)d_in[2], scaleBuf, shiftBuf, 64, 32000.f);
    bn_apply<<<dim3((32 * M0 + 255) / 256), dim3(256), 0, stream>>>(
        raw0, nullptr, feat0, scaleBuf, shiftBuf, 32, M0, 1);

    // ---- stage 1 ----
    fps_kernel<<<dim3(16), dim3(256), 0, stream>>>(x, sidx1, nxy1, 2000, 1000);
    zeroF(bstats, 32);
    knn_group<<<dim3(16000), dim3(256), 0, stream>>>(x, nxy1, feat0, cent, bstats,
                                                     2000, 1000, 64, M0, Mg1);
    bstd_fin<<<dim3(1), dim3(32), 0, stream>>>(bstats, rstdB, (float)(1000 * 64 * 66));
    {
        size_t tot = (size_t)80 * Mg1;
        group_scale<<<dim3((unsigned)((tot + 255) / 256)), dim3(256), 0, stream>>>(
            cent, rstdB, (const float*)d_in[5], (const float*)d_in[6],
            feat0, sidx1, Xg, 64, 80, Mg1, 1000, M0, 2000);
    }
    conv(wpT1, Xg, bufRaw, 128, 5, Mg1, 41, 40, nullptr, bufX, 1);
    resblock(bufX, bufRaw, bufH, 128, 4, Mg1, 23, 27, wpS0[4], wpS0[5]);
    resblock(bufX, bufRaw, bufH, 128, 4, Mg1, 31, 35, wpS0[6], wpS0[7]);
    maxpool_kernel<<<dim3((64 * M1 + 255) / 256), dim3(256), 0, stream>>>(bufX, p1, 64, M1);
    resblock(p1, posRaw, posH, 128, 4, M1, 7, 11, wpS0[0], wpS0[1]);
    resblock(p1, posRaw, posH, 128, 4, M1, 15, 19, wpS0[2], wpS0[3]);

    // ---- stage 2 ----
    fps_kernel<<<dim3(16), dim3(256), 0, stream>>>(nxy1, sidx2, nxy2, 1000, 500);
    zeroF(bstats, 32);
    knn_group<<<dim3(8000), dim3(256), 0, stream>>>(nxy1, nxy2, p1, cent, bstats,
                                                    1000, 500, 128, M1, Mg2);
    bstd_fin<<<dim3(1), dim3(32), 0, stream>>>(bstats, rstdB, (float)(500 * 64 * 130));
    {
        size_t tot = (size_t)144 * Mg2;
        group_scale<<<dim3((unsigned)((tot + 255) / 256)), dim3(256), 0, stream>>>(
            cent, rstdB, (const float*)d_in[43], (const float*)d_in[44],
            p1, sidx2, Xg, 128, 144, Mg2, 500, M1, 1000);
    }
    conv(wpT2, Xg, bufRaw, 256, 9, Mg2, 79, 78, nullptr, bufX, 1);
    resblock(bufX, bufRaw, bufH, 256, 8, Mg2, 61, 65, wpS1[4], wpS1[5]);
    resblock(bufX, bufRaw, bufH, 256, 8, Mg2, 69, 73, wpS1[6], wpS1[7]);
    maxpool_kernel<<<dim3((128 * M2 + 255) / 256), dim3(256), 0, stream>>>(bufX, p2, 128, M2);
    resblock(p2, posRaw, posH, 256, 8, M2, 45, 49, wpS1[0], wpS1[1]);
    resblock(p2, posRaw, posH, 256, 8, M2, 53, 57, wpS1[2], wpS1[3]);

    unpack_out<<<dim3((16 * 256 * 500 + 255) / 256), dim3(256), 0, stream>>>(p2, (float*)d_out);
}
